// TemporalGraphBatch_9594956939717
// MI455X (gfx1250) — matrix operand order repaired and checked by reading, not compiled, not run on hardware
//
#include <hip/hip_runtime.h>

// CDNA5 / gfx1250: wave32, WMMA 16x16x32 bf16, fp32 accumulate.
typedef __bf16 v16bf __attribute__((ext_vector_type(16)));
typedef float  v8f   __attribute__((ext_vector_type(8)));

#define T_DIM   100
#define A_DIM   19
#define NBATCH  16
#define FIN     512
#define FOUT    512
#define M_TOTAL (NBATCH * T_DIM * A_DIM)   // 30400 output rows
#define BM      128
#define BN      128
#define KC      32                          // bf16 WMMA K per step
#define NKC     (FIN / KC)                  // 16 K-chunks

// adj[pn-1, :] for the star graph (dis = deg^-1/2, deg_last = 1 + sum exp(-k)):
//   c[i<4] = exp(-(4-i)) * dis_last,  c[4] = dis_last^2
__device__ __constant__ float d_cw[5] = {
    0.01461134f, 0.03971775f, 0.10796401f, 0.29347659f, 0.63640869f
};

// A-matrix fragment (16-bit 16x32, this code's contiguous k order): lane (m = L%16, h = L/16)
// holds K = 16h..16h+15  -> two b128 LDS loads per fragment.
__device__ __forceinline__ v16bf load_fragA(const __bf16* s, int row, int h) {
    const uint4* q = reinterpret_cast<const uint4*>(s) + row * 4; // 32 bf16/row
    union { v16bf v; uint4 q[2]; } u;
    u.q[0] = q[2 * h];
    u.q[1] = q[2 * h + 1];
    return u.v;
}
// B-matrix fragment: lane (n = L%16, h = L/16) holds K = 16h..16h+15
// (contiguous per lane-half, same k order as load_fragA).
__device__ __forceinline__ v16bf load_fragB(const __bf16* s, int col, int h) {
    const uint4* q = reinterpret_cast<const uint4*>(s) + col * 4 + 2 * h;
    union { v16bf v; uint4 q[2]; } u;
    u.q[0] = q[0];
    u.q[1] = q[1];
    return u.v;
}

__global__ __launch_bounds__(256)
void tgb_fused_blend_gemm(const float* __restrict__ SF,
                          const float* __restrict__ W,
                          const float* __restrict__ bias,
                          float* __restrict__ out)
{
    __shared__ alignas(16) __bf16 As_hi[BM * KC];
    __shared__ alignas(16) __bf16 As_lo[BM * KC];
    __shared__ alignas(16) __bf16 Bs_hi[BN * KC];
    __shared__ alignas(16) __bf16 Bs_lo[BN * KC];

    const int tid = threadIdx.x;
    const int m0  = blockIdx.y * BM;
    const int n0  = blockIdx.x * BN;

    // ---- A-staging assignment: one (row, 16-float half) per thread ----
    const int a_row  = tid >> 1;
    const int a_half = tid & 1;
    int m_stage = m0 + a_row;
    if (m_stage >= M_TOTAL) m_stage = M_TOTAL - 1;          // clamp loads only
    const int t_stage = (m_stage % (T_DIM * A_DIM)) / A_DIM;
    int srcOff[5];
    #pragma unroll
    for (int n = 0; n < 5; ++n) {
        const int tt = t_stage + n;
        // window index t+n if in range, else the "filled" copy of t=0
        const int srow = (tt < T_DIM) ? (m_stage + n * A_DIM)
                                      : (m_stage - t_stage * A_DIM);
        srcOff[n] = srow * FIN + a_half * 16;
    }

    // ---- B-staging assignment: one (col, 16-float half) per thread ----
    const int wOff = (n0 + (tid >> 1)) * FIN + (tid & 1) * 16;

    // ---- wave tiling: 8 waves, each owns a 32(M) x 64(N) C strip ----
    const int lane = tid & 31;
    const int wid  = tid >> 5;
    const int wm   = wid & 3;      // M strip (x32 rows)
    const int wn   = wid >> 2;     // N strip (x64 cols)
    const int ln   = lane & 15;
    const int lh   = lane >> 4;

    v8f acc[2][4];
    #pragma unroll
    for (int mt = 0; mt < 2; ++mt)
        #pragma unroll
        for (int nt = 0; nt < 4; ++nt)
            acc[mt][nt] = (v8f){0.f, 0.f, 0.f, 0.f, 0.f, 0.f, 0.f, 0.f};

    for (int kc = 0; kc < NKC; ++kc) {
        const int koff = kc * KC;

        // ---- stage A: fused 5-tap temporal blend -> bf16 hi/lo split ----
        {
            float y[16];
            #pragma unroll
            for (int j = 0; j < 16; ++j) y[j] = 0.f;
            #pragma unroll
            for (int n = 0; n < 5; ++n) {
                const float c = d_cw[n];
                const float4* p = reinterpret_cast<const float4*>(SF + srcOff[n] + koff);
                #pragma unroll
                for (int j = 0; j < 4; ++j) {
                    const float4 v = p[j];
                    y[4 * j + 0] = fmaf(c, v.x, y[4 * j + 0]);
                    y[4 * j + 1] = fmaf(c, v.y, y[4 * j + 1]);
                    y[4 * j + 2] = fmaf(c, v.z, y[4 * j + 2]);
                    y[4 * j + 3] = fmaf(c, v.w, y[4 * j + 3]);
                }
            }
            union { __bf16 h[16]; uint4 q[2]; } uh, ul;
            #pragma unroll
            for (int j = 0; j < 16; ++j) {
                const float f   = y[j];
                const __bf16 hb = (__bf16)f;
                uh.h[j] = hb;
                ul.h[j] = (__bf16)(f - (float)hb);
            }
            uint4* dh = reinterpret_cast<uint4*>(As_hi) + a_row * 4 + a_half * 2;
            uint4* dl = reinterpret_cast<uint4*>(As_lo) + a_row * 4 + a_half * 2;
            dh[0] = uh.q[0]; dh[1] = uh.q[1];
            dl[0] = ul.q[0]; dl[1] = ul.q[1];
        }

        // ---- stage B: W tile -> bf16 hi/lo split ----
        {
            const float4* p = reinterpret_cast<const float4*>(W + wOff + koff);
            union { __bf16 h[16]; uint4 q[2]; } uh, ul;
            #pragma unroll
            for (int j = 0; j < 4; ++j) {
                const float4 v = p[j];
                const float f[4] = {v.x, v.y, v.z, v.w};
                #pragma unroll
                for (int k = 0; k < 4; ++k) {
                    const __bf16 hb = (__bf16)f[k];
                    uh.h[4 * j + k] = hb;
                    ul.h[4 * j + k] = (__bf16)(f[k] - (float)hb);
                }
            }
            uint4* dh = reinterpret_cast<uint4*>(Bs_hi) + (tid >> 1) * 4 + (tid & 1) * 2;
            uint4* dl = reinterpret_cast<uint4*>(Bs_lo) + (tid >> 1) * 4 + (tid & 1) * 2;
            dh[0] = uh.q[0]; dh[1] = uh.q[1];
            dl[0] = ul.q[0]; dl[1] = ul.q[1];
        }

        if (kc + 1 < NKC) { // lowers to global_prefetch_b8 (gfx1250 path)
            __builtin_prefetch(SF + srcOff[4] + koff + KC, 0, 1);
            __builtin_prefetch(W + wOff + koff + KC, 0, 1);
        }

        __syncthreads();

        // ---- WMMA: D = A*B + C, bf16 x bf16 -> f32, hi/lo split (3 products) ----
        v16bf ah[2], al[2];
        #pragma unroll
        for (int mt = 0; mt < 2; ++mt) {
            const int r = wm * 32 + mt * 16 + ln;
            ah[mt] = load_fragA(As_hi, r, lh);
            al[mt] = load_fragA(As_lo, r, lh);
        }
        #pragma unroll
        for (int nt = 0; nt < 4; ++nt) {
            const int cN = wn * 64 + nt * 16 + ln;
            const v16bf bh = load_fragB(Bs_hi, cN, lh);
            const v16bf bl = load_fragB(Bs_lo, cN, lh);
            #pragma unroll
            for (int mt = 0; mt < 2; ++mt) {
                acc[mt][nt] = __builtin_amdgcn_wmma_f32_16x16x32_bf16(
                    false, ah[mt], false, bh, (short)0, acc[mt][nt], false, false);
                acc[mt][nt] = __builtin_amdgcn_wmma_f32_16x16x32_bf16(
                    false, ah[mt], false, bl, (short)0, acc[mt][nt], false, false);
                acc[mt][nt] = __builtin_amdgcn_wmma_f32_16x16x32_bf16(
                    false, al[mt], false, bh, (short)0, acc[mt][nt], false, false);
            }
        }

        __syncthreads();
    }

    // ---- epilogue: bias + store. C layout: VGPR r -> M = r + 8*(lane/16), N = lane%16 ----
    #pragma unroll
    for (int nt = 0; nt < 4; ++nt) {
        const int n  = n0 + wn * 64 + nt * 16 + ln;
        const float bv = bias[n];
        #pragma unroll
        for (int mt = 0; mt < 2; ++mt) {
            const int mbase = m0 + wm * 32 + mt * 16 + 8 * lh;
            #pragma unroll
            for (int r = 0; r < 8; ++r) {
                const int mg = mbase + r;
                if (mg < M_TOTAL)
                    out[(size_t)mg * FOUT + n] = acc[mt][nt][r] + bv;
            }
        }
    }
}

extern "C" void kernel_launch(void* const* d_in, const int* in_sizes, int n_in,
                              void* d_out, int out_size, void* d_ws, size_t ws_size,
                              hipStream_t stream) {
    (void)in_sizes; (void)n_in; (void)out_size; (void)d_ws; (void)ws_size;
    const float* SF = (const float*)d_in[0];   // (16,100,19,512) fp32
    const float* W  = (const float*)d_in[1];   // (512,512) fp32
    const float* b  = (const float*)d_in[2];   // (512,) fp32
    float* out = (float*)d_out;                // (16,100,19,512) fp32

    dim3 grid(FOUT / BN, (M_TOTAL + BM - 1) / BM);  // 4 x 238
    tgb_fused_blend_gemm<<<grid, dim3(256), 0, stream>>>(SF, W, b, out);
}
